// VQVAE_84610855731805
// MI455X (gfx1250) — compile-verified, matrix operand order checked by reading, not run on hardware
//
#include <hip/hip_runtime.h>

// ---------------------------------------------------------------------------
// VQ-VAE nearest-codebook lookup for gfx1250 (MI455X), v3.
//   argmin_k ||z-c_k||^2  ==  argmin_k ( c_sq[k] - 2 z.c_k )   (z_sq constant)
// z.c via fp16-split WMMA: z=zh+zl, c=ch+cl, z.c ~= zh.ch + zh.cl + zl.ch
// (fp32 accumulate -> ~fp32-accurate argmin) with v_wmma_f32_16x16x32_f16.
//
// v3: double-buffered LDS staging of the codebook (64 codes/group, hi+lo)
// via global_load_async_to_lds_b128; prefetch of group g+1 overlaps compute
// of group g (ASYNCcnt<=32 ordering). A fragments are either pre-split f16
// from workspace (if ws_size allows) or split on the fly from fp32 z_e.
// ---------------------------------------------------------------------------

typedef _Float16 v16h __attribute__((ext_vector_type(16)));
typedef _Float16 h8   __attribute__((ext_vector_type(8)));
typedef float    v8f  __attribute__((ext_vector_type(8)));

#define N_TOK 32768
#define DIM   512
#define KCB   4096
#define BM    128                 // rows per block (8 waves x 16 rows)
#define GCODES 64                 // codes staged per group (per buffer)
#define NGROUP (KCB / GCODES)     // 64
#define NTILE  (GCODES / 16)      // 4 column tiles per group
#define LSTR  520                 // padded halves per LDS row: 1040B stride =
                                  // 260 dwords == 4 (mod 64) -> conflict-free
#define BUFH  (GCODES * LSTR)     // halves per array (hi or lo) per buffer

union HFrag { v16h v; h8 h[2]; };

// Fragment per ISA 7.12.2 (16-bit 16x32): lane = (h<<4)|m,
// halves[0..7] <- K = o0..o0+7, halves[8..15] <- K = o0+16..o0+23,
// where o0 = kc + h*8 (caller folds h in).
__device__ __forceinline__ v16h load_frag16(const _Float16* p, int o0) {
  HFrag f;
  f.h[0] = *(const h8*)(p + o0);
  f.h[1] = *(const h8*)(p + o0 + 16);
  return f.v;
}

// fp32 -> f16 hi/lo split of one A fragment (two 8-float runs at p, p+16)
__device__ __forceinline__ void split_fragA(const float* p, v16h& hi, v16h& lo) {
#pragma unroll
  for (int j = 0; j < 8; ++j) {
    float v0 = p[j];
    _Float16 h0 = (_Float16)v0;
    hi[j] = h0;
    lo[j] = (_Float16)(v0 - (float)h0);
    float v1 = p[16 + j];
    _Float16 h1 = (_Float16)v1;
    hi[8 + j] = h1;
    lo[8 + j] = (_Float16)(v1 - (float)h1);
  }
}

// Async copy of one 16B chunk: LDS[lds_off] <= *(gbase + goff)   [ASYNCcnt]
__device__ __forceinline__ void async_copy_b128(unsigned lds_off,
                                                const void* gbase,
                                                unsigned goff) {
  asm volatile("global_load_async_to_lds_b128 %0, %1, %2"
               :
               : "v"(lds_off), "v"(goff), "s"(gbase)
               : "memory");
}

template <int N>
__device__ __forceinline__ void wait_async_le() {
  asm volatile("s_wait_asynccnt %0" ::"i"(N) : "memory");
}

// stage one 64-code group (hi+lo) into LDS buffer `buf` (32 asyncs/thread)
__device__ __forceinline__ void stage_group(const _Float16* ch,
                                            const _Float16* cl,
                                            int g, int buf, int tid) {
  const unsigned base = (unsigned)buf * (2u * BUFH) * 2u;   // bytes
  for (int c = tid; c < GCODES * (DIM / 8); c += 256) {
    int row   = c >> 6;                                     // 0..63
    int chunk = c & 63;                                     // 16B chunk in row
    unsigned lds_off = base + (unsigned)(row * LSTR + chunk * 8) * 2u;
    unsigned goff = ((unsigned)(g * GCODES + row) * DIM + chunk * 8) * 2u;
    async_copy_b128(lds_off, ch, goff);                     // hi
    async_copy_b128(lds_off + BUFH * 2u, cl, goff);         // lo
  }
}

__global__ void vq_zero_freq(float* freq) {
  int i = blockIdx.x * blockDim.x + threadIdx.x;
  if (i < KCB) freq[i] = 0.0f;
}

// codebook fp32 -> f16 hi/lo split + per-code squared norm (fp32)
__global__ void vq_prep(const float* __restrict__ cb,
                        _Float16* __restrict__ ch, _Float16* __restrict__ cl,
                        float* __restrict__ csq) {
  __shared__ float red[256];
  const int k = blockIdx.x, t = threadIdx.x;
  const float* row = cb + (size_t)k * DIM;
  float acc = 0.0f;
  for (int i = t; i < DIM; i += 256) {
    float v = row[i];
    _Float16 h = (_Float16)v;
    ch[(size_t)k * DIM + i] = h;
    cl[(size_t)k * DIM + i] = (_Float16)(v - (float)h);
    acc += v * v;
  }
  red[t] = acc;
  __syncthreads();
  for (int s = 128; s > 0; s >>= 1) {
    if (t < s) red[t] += red[t + s];
    __syncthreads();
  }
  if (t == 0) csq[k] = red[0];
}

// z fp32 -> f16 hi/lo split (one element per thread)
__global__ void vq_prep_z(const float* __restrict__ z,
                          _Float16* __restrict__ zh, _Float16* __restrict__ zl) {
  size_t i = (size_t)blockIdx.x * blockDim.x + threadIdx.x;
  float v = z[i];
  _Float16 h = (_Float16)v;
  zh[i] = h;
  zl[i] = (_Float16)(v - (float)h);
}

template <bool F16Z>
__global__ __launch_bounds__(256) void vq_main(
    const float* __restrict__ z_e, const float* __restrict__ codebook,
    const _Float16* __restrict__ ch, const _Float16* __restrict__ cl,
    const float* __restrict__ csq,
    const _Float16* __restrict__ zh_g, const _Float16* __restrict__ zl_g,
    float* __restrict__ zq, float* __restrict__ out_idx,
    float* __restrict__ freq) {
  extern __shared__ _Float16 smem[];   // 2 buffers x { hi[64][LSTR], lo[...] }

  const int tid  = threadIdx.x;
  const int lane = tid & 31;
  const int wave = tid >> 5;
  const int blk  = blockIdx.x;
  const int m16  = lane & 15;
  const int hsel = lane >> 4;

  const size_t rowIdx = (size_t)(blk * BM + wave * 16 + m16);
  const float*    zrow  = z_e  + rowIdx * DIM;   // fp32 path
  const _Float16* zhrow = F16Z ? zh_g + rowIdx * DIM : nullptr;
  const _Float16* zlrow = F16Z ? zl_g + rowIdx * DIM : nullptr;

  float vmin[8];
  int   vidx[8];
#pragma unroll
  for (int r = 0; r < 8; ++r) { vmin[r] = 3.4e38f; vidx[r] = 0; }

  stage_group(ch, cl, 0, 0, tid);

  for (int g = 0; g < NGROUP; ++g) {
    const int buf = g & 1;
    if (g + 1 < NGROUP) {
      stage_group(ch, cl, g + 1, buf ^ 1, tid);   // prefetch next group
      wait_async_le<32>();                        // in-order: group g landed
    } else {
      wait_async_le<0>();
    }
    __syncthreads();

    const _Float16* bufbase = smem + buf * (2 * BUFH);

    v8f acc[NTILE];
#pragma unroll
    for (int t = 0; t < NTILE; ++t) acc[t] = (v8f){};

#pragma unroll 2
    for (int kc = 0; kc < DIM; kc += 32) {
      const int o0 = kc + hsel * 8;
      v16h ah, al;
      if (F16Z) {
        ah = load_frag16(zhrow, o0);
        al = load_frag16(zlrow, o0);
      } else {
        split_fragA(zrow + o0, ah, al);
      }
#pragma unroll
      for (int t = 0; t < NTILE; ++t) {
        const _Float16* bb = bufbase + (t * 16 + m16) * LSTR;
        v16h bh = load_frag16(bb, o0);
        v16h bl = load_frag16(bb + BUFH, o0);
        acc[t] = __builtin_amdgcn_wmma_f32_16x16x32_f16(
            false, ah, false, bh, (short)0, acc[t], false, false);
        acc[t] = __builtin_amdgcn_wmma_f32_16x16x32_f16(
            false, ah, false, bl, (short)0, acc[t], false, false);
        acc[t] = __builtin_amdgcn_wmma_f32_16x16x32_f16(
            false, al, false, bh, (short)0, acc[t], false, false);
      }
    }

    // running argmin update (score = c_sq - 2*dot; z_sq irrelevant)
#pragma unroll
    for (int t = 0; t < NTILE; ++t) {
      const int code = g * GCODES + t * 16 + m16;
      const float cs = csq[code];
#pragma unroll
      for (int r = 0; r < 8; ++r) {
        float s = cs - 2.0f * acc[t][r];
        if (s < vmin[r]) { vmin[r] = s; vidx[r] = code; }
      }
    }

    __syncthreads();   // all consumers done before buffer `buf` is re-staged
  }

  // ---- xor-butterfly argmin across the 16 column-lanes of each half
  // (masks 1,2,4,8 never cross the h boundary); smallest index wins ties.
#pragma unroll
  for (int r = 0; r < 8; ++r) {
    float v = vmin[r];
    int  ix = vidx[r];
#pragma unroll
    for (int s = 8; s >= 1; s >>= 1) {
      float v2 = __shfl_xor(v, s, 32);
      int   i2 = __shfl_xor(ix, s, 32);
      if (v2 < v || (v2 == v && i2 < ix)) { v = v2; ix = i2; }
    }
    vmin[r] = v;
    vidx[r] = ix;
  }
  // every lane in half h now holds the result for row (r + 8*h)

  // ---- outputs: index, freq histogram, z_q gather (wave-wide float4 copy)
  const int rowBase = blk * BM + wave * 16;
#pragma unroll
  for (int m = 0; m < 16; ++m) {
    const int idx = __shfl(vidx[m & 7], (m >> 3) * 16, 32);  // uniform
    const int row = rowBase + m;
    if (lane == m) {
      out_idx[row] = (float)idx;
      atomicAdd(&freq[idx], 1.0f);
    }
    const float4* src = (const float4*)(codebook + (size_t)idx * DIM);
    float4* dst = (float4*)(zq + (size_t)row * DIM);
#pragma unroll
    for (int j = 0; j < DIM / 4 / 32; ++j)
      dst[lane + j * 32] = src[lane + j * 32];
  }
}

extern "C" void kernel_launch(void* const* d_in, const int* in_sizes, int n_in,
                              void* d_out, int out_size, void* d_ws, size_t ws_size,
                              hipStream_t stream) {
  const float* z_e      = (const float*)d_in[0];
  const float* codebook = (const float*)d_in[1];
  // d_in[2] = is_train: forward value of z_q is identical either way; ignored.

  float* out     = (float*)d_out;
  float* zq      = out;                          // [N, D]
  float* out_idx = out + (size_t)N_TOK * DIM;    // [N]
  float* freq    = out_idx + N_TOK;              // [K]

  _Float16* ch = (_Float16*)d_ws;                  // [K, D] f16 hi   (4 MB)
  _Float16* cl = ch + (size_t)KCB * DIM;           // [K, D] f16 lo   (4 MB)
  float*   csq = (float*)(cl + (size_t)KCB * DIM); // [K]             (16 KB)
  _Float16* zh = (_Float16*)(csq + KCB);           // [N, D] f16 hi   (32 MB)
  _Float16* zl = zh + (size_t)N_TOK * DIM;         // [N, D] f16 lo   (32 MB)

  const size_t need_f16z =
      (size_t)KCB * DIM * 4 + (size_t)KCB * 4 + (size_t)N_TOK * DIM * 4;
  const bool f16z = ws_size >= need_f16z;

  vq_zero_freq<<<(KCB + 255) / 256, 256, 0, stream>>>(freq);
  vq_prep<<<KCB, 256, 0, stream>>>(codebook, ch, cl, csq);

  const size_t lds_bytes = (size_t)4 * GCODES * LSTR * sizeof(_Float16); // 260KB
  if (f16z) {
    vq_prep_z<<<(N_TOK * DIM) / 256, 256, 0, stream>>>(z_e, zh, zl);
    vq_main<true><<<N_TOK / BM, 256, lds_bytes, stream>>>(
        z_e, codebook, ch, cl, csq, zh, zl, zq, out_idx, freq);
  } else {
    vq_main<false><<<N_TOK / BM, 256, lds_bytes, stream>>>(
        z_e, codebook, ch, cl, csq, nullptr, nullptr, zq, out_idx, freq);
  }
}
